// GGAM_2362232013241
// MI455X (gfx1250) — matrix operand order repaired and checked by reading, not compiled, not run on hardware
//
#include <hip/hip_runtime.h>
#include <hip/hip_bf16.h>
#include <cstdint>

typedef __attribute__((ext_vector_type(16))) _Float16 v16h;
typedef __attribute__((ext_vector_type(8)))  float    v8f;

#define B_   2
#define C_   128
#define H_   48
#define W_   64
#define HW_  3072
#define KW_  41
#define RR_  20
#define KK_  1681
#define CH_  32
#define NT_  192                 // 16-wide tiles per matrix dim
#define SCALE_QK 0.08838834764831845f
#define DM_SC_ 100.0f

// ---- workspace layout (bytes, all 256-aligned) ----
#define OFF_EXPY   0u            // 48*48 f32
#define OFF_EXPX   9216u         // 64*64 f32
#define OFF_MEAN   25600u        // B*HW f32
#define OFF_CRN    50176u        // B*HW int4  (bilinear corner indices)
#define OFF_WGT    148480u       // B*HW float2 (bilinear weights)
#define OFF_GA     197632u       // B*HW f32
#define OFF_MX     222208u       // B*HW f32 softmax max
#define OFF_IDN    246784u       // B*HW f32 softmax 1/denom
#define OFF_QF     271360u       // B*C*HW f32
#define OFF_KF     3417088u      // B*C*HW f32
#define OFF_QH     6562816u      // swizzled f16 (B operand)
#define OFF_KH     8135680u      // swizzled f16 (A operand)
#define OFF_FVT    9708544u      // B*HW*C f32 (v, transposed)
#define OFF_AT     12854272u     // B*HW*HW f32 : At[n(query)][m(src)]

__device__ __forceinline__ float thrG(float e) {
  float v = fmaf(e, 2.5f, 10.0f);          // g*SC_G + BI
  return v < 10.25f ? 0.0f : v;            // threshold BI + THR*SC_G
}

// ---- separable gaussian exp tables ----
__global__ void k_tables(float* expY, float* expX) {
  int i = blockIdx.x * 256 + threadIdx.x;
  if (i < 48 * 48) {
    float d = (float)(i / 48 - i % 48);
    expY[i] = __expf(-(d * d) * (1.0f / 800.0f));   // 2*sigma^2 = 800
  }
  int j = i - 48 * 48;
  if (j >= 0 && j < 64 * 64) {
    float d = (float)(j / 64 - j % 64);
    expX[j] = __expf(-(d * d) * (1.0f / 800.0f));
  }
}

// ---- channel mean of feat_mo ----
__global__ void k_mean(const float* __restrict__ fmo, float* __restrict__ mean) {
  int t = blockIdx.x * 256 + threadIdx.x;
  if (t >= B_ * HW_) return;
  int bb = t / HW_, n = t % HW_;
  const float* p = fmo + (size_t)bb * C_ * HW_ + n;
  float s = 0.f;
  for (int c = 0; c < C_; ++c) s += p[(size_t)c * HW_];
  mean[t] = s * (1.0f / C_);
}

// ---- DGM: depthwise conv -> offsets -> grid-sample corner data + gate ga ----
__global__ void k_dgm(const float* __restrict__ mean,
                      const float* __restrict__ w_in, const float* __restrict__ b_in,
                      const float* __restrict__ w_dw, const float* __restrict__ b_dw,
                      const float* __restrict__ w_ofs, const float* __restrict__ w_a,
                      const float* __restrict__ b_a, const float* __restrict__ lam,
                      int4* __restrict__ crn, float2* __restrict__ wgt,
                      float* __restrict__ ga) {
  int t = blockIdx.x * 128 + threadIdx.x;
  if (t >= B_ * HW_) return;
  int bb = t / HW_, n = t % HW_;
  int y = n / W_, x = n % W_;
  float fm[25];
  for (int ky = 0; ky < 5; ++ky) {
    int yy = min(max(y + ky - 2, 0), H_ - 1);
    for (int kx = 0; kx < 5; ++kx) {
      int xx = min(max(x + kx - 2, 0), W_ - 1);
      fm[ky * 5 + kx] = mean[bb * HW_ + yy * W_ + xx];   // replicate pad
    }
  }
  float o0 = 0.f, o1 = 0.f, fa = 0.f;
  for (int ch = 0; ch < CH_; ++ch) {
    float win = w_in[ch], bin = b_in[ch];
    float s = b_dw[ch];
    for (int u = 0; u < 25; ++u)
      s = fmaf(w_dw[ch * 25 + u], fmaf(fm[u], win, bin), s);
    float d = fmaxf(s, 0.f);
    o0 = fmaf(w_ofs[ch], d, o0);
    o1 = fmaf(w_ofs[CH_ + ch], d, o1);
    fa = fmaf(w_a[ch], fmaf(fm[12], win, bin), fa);
  }
  float oy = tanhf(o0) * (DM_SC_ / H_);
  float ox = tanhf(o1) * (DM_SC_ / W_);
  float ry = ((y + 0.5f) / H_) * 2.f - 1.f;
  float rx = ((x + 0.5f) / W_) * 2.f - 1.f;
  float py = fminf(fmaxf(oy + ry, -1.f), 1.f);
  float px = fminf(fmaxf(ox + rx, -1.f), 1.f);
  float sy = (py + 1.f) * 0.5f * (H_ - 1);               // align_corners=True
  float sx = (px + 1.f) * 0.5f * (W_ - 1);
  float y0f = floorf(sy), x0f = floorf(sx);
  int y0 = min(max((int)y0f, 0), H_ - 1);
  int y1 = min(max((int)y0f + 1, 0), H_ - 1);
  int x0 = min(max((int)x0f, 0), W_ - 1);
  int x1 = min(max((int)x0f + 1, 0), W_ - 1);
  crn[t] = make_int4(y0, y1, x0, x1);
  wgt[t] = make_float2(sy - y0f, sx - x0f);
  ga[t]  = 1.f + lam[0] * (fa + b_a[0]);
}

// ---- q/k 1x1 conv (scale folded into q) ----
__global__ void k_qk(const float* __restrict__ ctx, const float* __restrict__ w_qk,
                     float* __restrict__ qf, float* __restrict__ kf) {
  int t = blockIdx.x * 256 + threadIdx.x;
  int n = t % HW_;
  int r = t / HW_;
  int c2 = r % 256;
  int bb = r / 256;
  const float* cp = ctx + (size_t)bb * C_ * HW_ + n;
  const float* wp = w_qk + c2 * C_;
  float s = 0.f;
  for (int cc = 0; cc < C_; ++cc) s = fmaf(wp[cc], cp[(size_t)cc * HW_], s);
  if (c2 < C_) qf[((size_t)bb * C_ + c2) * HW_ + n] = s * SCALE_QK;
  else         kf[((size_t)bb * C_ + (c2 - C_)) * HW_ + n] = s;
}

// ---- pack q (B operand) / k (A operand) into WMMA per-lane f16 layouts ----
// A 16x32 f16 (ISA 7.12.2):  lane grp0: halves j<8->K=j, j>=8->K=j+8 ; grp1: +8
// B 32x16 f16 (same K order as A):  lane grp0: halves j<8->K=j, j>=8->K=j+8 ; grp1: +8
__global__ void k_pack(const float* __restrict__ qf, const float* __restrict__ kf,
                       _Float16* __restrict__ qh, _Float16* __restrict__ kh) {
  int t = blockIdx.x * 256 + threadIdx.x;
  int j = t & 15;  int e = t >> 4;
  int lane = e & 31; e >>= 5;
  int kc = e & 3;  e >>= 2;
  int tile = e % NT_; e /= NT_;
  int bb = e & 1;  int mat = e >> 1;
  int col = tile * 16 + (lane & 15);
  int grp = lane >> 4;
  size_t dst = ((((size_t)bb * NT_ + tile) * 4 + kc) * 32 + lane) * 16 + j;
  if (mat == 0) {
    int K = j + 8 * grp + (j >= 8 ? 8 : 0);
    qh[dst] = (_Float16)qf[((size_t)bb * C_ + (kc * 32 + K)) * HW_ + col];
  } else {
    int K = j + 8 * grp + (j >= 8 ? 8 : 0);
    kh[dst] = (_Float16)kf[((size_t)bb * C_ + (kc * 32 + K)) * HW_ + col];
  }
}

// ---- v projection, stored transposed: fvT[b][m][c] ----
__global__ void k_fvT(const float* __restrict__ fmo, const float* __restrict__ w_v,
                      float* __restrict__ fvT) {
  int m = blockIdx.x * 256 + threadIdx.x;
  int c = blockIdx.y;
  int bb = blockIdx.z;
  const float* fp = fmo + (size_t)bb * C_ * HW_ + m;
  const float* wp = w_v + c * C_;
  float s = 0.f;
  for (int cc = 0; cc < C_; ++cc) s = fmaf(wp[cc], fp[(size_t)cc * HW_], s);
  fvT[((size_t)bb * HW_ + m) * C_ + c] = s;
}

// ---- WMMA attention GEMM, fused gd/ga epilogue, stores At[n(query)][m(src)] ----
__global__ void __launch_bounds__(128) k_attn(
    const _Float16* __restrict__ kh, const _Float16* __restrict__ qh,
    const float* __restrict__ expY, const float* __restrict__ expX,
    const int4* __restrict__ crn, const float2* __restrict__ wgt,
    const float* __restrict__ ga, float* __restrict__ At) {
  __shared__ float sY[48 * 48];
  __shared__ float sX[64 * 64];
  for (int i = threadIdx.x; i < 48 * 48; i += 128) sY[i] = expY[i];
  for (int i = threadIdx.x; i < 64 * 64; i += 128) sX[i] = expX[i];
  __syncthreads();

  int bb = blockIdx.z;
  int wave = threadIdx.x >> 5;
  int lane = threadIdx.x & 31;
  int wt = blockIdx.x * 4 + wave;            // 0..2303  (48x48 wave tiles)
  int m0 = (wt / 48) * 64;                   // src rows of 64x64 tile
  int n0 = (wt % 48) * 64;                   // query cols

  const v8f zero = {0.f, 0.f, 0.f, 0.f, 0.f, 0.f, 0.f, 0.f};
  v8f acc[4][4];
#pragma unroll
  for (int i = 0; i < 4; ++i)
#pragma unroll
    for (int j = 0; j < 4; ++j) acc[i][j] = zero;

  const v16h* Ab = (const v16h*)kh;
  const v16h* Bb = (const v16h*)qh;
#pragma unroll
  for (int kc = 0; kc < 4; ++kc) {
    v16h av[4], bv[4];
#pragma unroll
    for (int i = 0; i < 4; ++i)
      av[i] = Ab[(((size_t)bb * NT_ + (m0 >> 4) + i) * 4 + kc) * 32 + lane];
#pragma unroll
    for (int j = 0; j < 4; ++j)
      bv[j] = Bb[(((size_t)bb * NT_ + (n0 >> 4) + j) * 4 + kc) * 32 + lane];
#pragma unroll
    for (int i = 0; i < 4; ++i)
#pragma unroll
      for (int j = 0; j < 4; ++j)
        acc[i][j] = __builtin_amdgcn_wmma_f32_16x16x32_f16(
            false, av[i], false, bv[j], (short)0, acc[i][j], false, false);
  }

  // Epilogue: attn_g[m,n] = attn[m,n] * gd(chan=n, pos=m) * ga[m] -> At[n][m]
  int grp = lane >> 4;
  int l16 = lane & 15;
#pragma unroll
  for (int i = 0; i < 4; ++i) {
    int mb = m0 + 16 * i + 8 * grp;          // 8 consecutive m for this lane
    int4 cr[8]; float2 wg[8]; float gav[8];
#pragma unroll
    for (int r = 0; r < 8; ++r) {
      int idx = bb * HW_ + mb + r;
      cr[r] = crn[idx]; wg[r] = wgt[idx]; gav[r] = ga[idx];
    }
#pragma unroll
    for (int jj = 0; jj < 4; ++jj) {
      int n = n0 + 16 * jj + l16;
      int iy = n >> 6, ix = n & 63;
      float o[8];
#pragma unroll
      for (int r = 0; r < 8; ++r) {
        float eY0 = sY[cr[r].x * 48 + iy];
        float eY1 = sY[cr[r].y * 48 + iy];
        float eX0 = sX[cr[r].z * 64 + ix];
        float eX1 = sX[cr[r].w * 64 + ix];
        float g00 = thrG(eY0 * eX0), g01 = thrG(eY0 * eX1);
        float g10 = thrG(eY1 * eX0), g11 = thrG(eY1 * eX1);
        float wy = wg[r].x, wx = wg[r].y;
        float g = (1.f - wy) * ((1.f - wx) * g00 + wx * g01)
                +        wy  * ((1.f - wx) * g10 + wx * g11);
        o[r] = acc[i][jj][r] * g * gav[r];
      }
      float* row = At + ((size_t)(bb * HW_ + n)) * HW_ + mb;
      *(float4*)(row)     = make_float4(o[0], o[1], o[2], o[3]);
      *(float4*)(row + 4) = make_float4(o[4], o[5], o[6], o[7]);
    }
  }
}

// ---- per-query softmax statistics over its 41x41 window (zeros kept for OOB) ----
__global__ void __launch_bounds__(128) k_soft(const float* __restrict__ At,
                                              float* __restrict__ mxArr,
                                              float* __restrict__ idn) {
  __shared__ float buf[KK_];
  __shared__ float red[128];
  int q = blockIdx.x, bb = blockIdx.z;
  int yq = q >> 6, xq = q & 63;
  const float* row = At + ((size_t)(bb * HW_ + q)) * HW_;
  for (int k = threadIdx.x; k < KK_; k += 128) {
    int ym = yq + k / KW_ - RR_;
    int xm = xq + k % KW_ - RR_;
    bool valid = (ym >= 0) & (ym < H_) & (xm >= 0) & (xm < W_);
    buf[k] = valid ? row[ym * W_ + xm] : 0.f;   // masked-to-zero, kept in softmax
  }
  __syncthreads();
  float m = -3.0e38f;
  for (int k = threadIdx.x; k < KK_; k += 128) m = fmaxf(m, buf[k]);
  red[threadIdx.x] = m; __syncthreads();
  for (int s = 64; s > 0; s >>= 1) {
    if (threadIdx.x < s) red[threadIdx.x] = fmaxf(red[threadIdx.x], red[threadIdx.x + s]);
    __syncthreads();
  }
  float mx = red[0];
  __syncthreads();
  float sum = 0.f;
  for (int k = threadIdx.x; k < KK_; k += 128) sum += __expf(buf[k] - mx);
  red[threadIdx.x] = sum; __syncthreads();
  for (int s = 64; s > 0; s >>= 1) {
    if (threadIdx.x < s) red[threadIdx.x] += red[threadIdx.x + s];
    __syncthreads();
  }
  if (threadIdx.x == 0) {
    mxArr[bb * HW_ + q] = mx;
    idn[bb * HW_ + q] = 1.0f / red[0];
  }
}

// ---- transpose aggregation: out[c,s] = fmo + gamma * sum_q w(q,s) * fv[c,q] ----
__global__ void __launch_bounds__(256) k_agg(
    const float* __restrict__ At, const float* __restrict__ mxArr,
    const float* __restrict__ idn, const float* __restrict__ fvT,
    const float* __restrict__ fmo, const float* __restrict__ gamma,
    float* __restrict__ out) {
  __shared__ float wl[16][16];
  __shared__ int qIdxS[16];
  int bb = blockIdx.z;
  int s0 = blockIdx.x * 16;                  // 16 consecutive s, same row
  int ys = s0 >> 6, xs0 = s0 & 63;
  int qy0 = max(ys - RR_, 0), qy1 = min(ys + RR_, H_ - 1);
  int qx0 = max(xs0 - RR_, 0), qx1 = min(xs0 + 15 + RR_, W_ - 1);
  int nQx = qx1 - qx0 + 1;
  int nQ = (qy1 - qy0 + 1) * nQx;
  int tid = threadIdx.x;
  int c = tid & 127, sg = tid >> 7;          // this thread: channel c, s-group sg
  int qi = tid >> 4, sl = tid & 15;          // staging role: (query-in-chunk, s-lane)
  float accv[8];
#pragma unroll
  for (int u = 0; u < 8; ++u) accv[u] = 0.f;

  for (int base = 0; base < nQ; base += 16) {
    int qq = base + qi;
    float wv = 0.f;
    int qidx = -1;
    if (qq < nQ) {
      int qy = qy0 + qq / nQx;
      int qx = qx0 + qq % nQx;
      qidx = qy * W_ + qx;
      int dx = qx - (xs0 + sl); if (dx < 0) dx = -dx;
      if (dx <= RR_) {                       // s within q's window (y always is)
        int g = bb * HW_ + qidx;
        wv = __expf(At[((size_t)g) * HW_ + s0 + sl] - mxArr[g]) * idn[g];
      }
    }
    __syncthreads();
    wl[qi][sl] = wv;
    if (sl == 0) qIdxS[qi] = qidx;
    __syncthreads();
#pragma unroll 4
    for (int u16 = 0; u16 < 16; ++u16) {
      int qg = qIdxS[u16];
      int qgc = qg < 0 ? 0 : qg;
      float fvq = fvT[((size_t)(bb * HW_ + qgc)) * C_ + c];
      if (qg < 0) fvq = 0.f;                 // weights are zero anyway
#pragma unroll
      for (int u = 0; u < 8; ++u)
        accv[u] = fmaf(wl[u16][sg * 8 + u], fvq, accv[u]);
    }
  }
  float gm = gamma[0];
#pragma unroll
  for (int u = 0; u < 8; ++u) {
    size_t oi = ((size_t)bb * C_ + c) * HW_ + s0 + sg * 8 + u;
    out[oi] = fmaf(gm, accv[u], fmo[oi]);
  }
}

extern "C" void kernel_launch(void* const* d_in, const int* in_sizes, int n_in,
                              void* d_out, int out_size, void* d_ws, size_t ws_size,
                              hipStream_t stream) {
  const float* feat_ctx = (const float*)d_in[0];
  const float* feat_mo  = (const float*)d_in[1];
  // d_in[2] = itr (unused: itr==0 path)
  const float* w_qk  = (const float*)d_in[3];
  const float* w_v   = (const float*)d_in[4];
  const float* gamma = (const float*)d_in[5];
  const float* w_in  = (const float*)d_in[6];
  const float* b_in  = (const float*)d_in[7];
  const float* w_dw  = (const float*)d_in[8];
  const float* b_dw  = (const float*)d_in[9];
  const float* w_ofs = (const float*)d_in[10];
  const float* w_a   = (const float*)d_in[11];
  const float* b_a   = (const float*)d_in[12];
  const float* lam   = (const float*)d_in[13];
  float* out = (float*)d_out;
  char* ws = (char*)d_ws;

  float*    expY = (float*)(ws + OFF_EXPY);
  float*    expX = (float*)(ws + OFF_EXPX);
  float*    mean = (float*)(ws + OFF_MEAN);
  int4*     crn  = (int4*)(ws + OFF_CRN);
  float2*   wgt  = (float2*)(ws + OFF_WGT);
  float*    ga   = (float*)(ws + OFF_GA);
  float*    mx   = (float*)(ws + OFF_MX);
  float*    idn  = (float*)(ws + OFF_IDN);
  float*    qf   = (float*)(ws + OFF_QF);
  float*    kf   = (float*)(ws + OFF_KF);
  _Float16* qh   = (_Float16*)(ws + OFF_QH);
  _Float16* kh   = (_Float16*)(ws + OFF_KH);
  float*    fvT  = (float*)(ws + OFF_FVT);
  float*    At   = (float*)(ws + OFF_AT);

  k_tables<<<25, 256, 0, stream>>>(expY, expX);
  k_mean<<<24, 256, 0, stream>>>(feat_mo, mean);
  k_dgm<<<48, 128, 0, stream>>>(mean, w_in, b_in, w_dw, b_dw, w_ofs, w_a, b_a,
                                lam, crn, wgt, ga);
  k_qk<<<6144, 256, 0, stream>>>(feat_ctx, w_qk, qf, kf);
  k_pack<<<6144, 256, 0, stream>>>(qf, kf, qh, kh);
  k_fvT<<<dim3(12, 128, 2), 256, 0, stream>>>(feat_mo, w_v, fvT);
  k_attn<<<dim3(576, 1, 2), 128, 0, stream>>>(kh, qh, expY, expX, crn, wgt, ga, At);
  k_soft<<<dim3(3072, 1, 2), 128, 0, stream>>>(At, mx, idn);
  k_agg<<<dim3(192, 1, 2), 256, 0, stream>>>(At, mx, idn, fvT, feat_mo, gamma, out);
}
